// ExtendedISAModule_23347442221063
// MI455X (gfx1250) — matrix operand order repaired and checked by reading, not compiled, not run on hardware
//
#include <hip/hip_runtime.h>

typedef __bf16 bf16_t;
typedef __bf16 v16bf __attribute__((ext_vector_type(16)));
typedef __bf16 v8bf  __attribute__((ext_vector_type(8)));
typedef float  v8f   __attribute__((ext_vector_type(8)));
typedef float  v4f   __attribute__((ext_vector_type(4)));

// ---- B-operand source (always [N,K] "B-transposed" layout, contiguous along K)
enum { B_BF16_NK = 0,  // bf16 [N,K] row-major: direct vector copy
       B_F32_NK  = 1 };// fp32 [N,K] row-major (weights): cvt->bf16 while staging
// ---- epilogue modes ----
enum { EPI_BF16 = 0,            // bf16 store, row-major [M,N]
       EPI_BF16T = 1,           // bf16 store TRANSPOSED [N,M] (16B chunk/lane)
       EPI_F32 = 2,             // fp32 store
       EPI_RELU_BIAS_BF16 = 3,  // relu(x + bias[col]) -> bf16
       EPI_RES_DUAL = 4,        // x + res -> fp32 Co AND bf16 Cb
       EPI_BIAS_RES_DUAL = 5 }; // x + bias[col] + res -> fp32 Co AND bf16 Cb

// C[M,N] = sum_h A_h (M x K, bf16 row-major) * B_h^T (B stored [N,K]).
// All contraction reads are contiguous along K: LDS staging is 100%
// vectorized b128 stores. 128x128 block tile, K-step 32, double-buffered
// LDS (1 barrier/step), 8 wave32; each wave 32x64 -> 8x
// v_wmma_f32_16x16x32_bf16 per K-step (f32 accumulate).
template<int BMODE, int EPI>
__global__ __launch_bounds__(256)
void gemm_wmma_bf16(const bf16_t* __restrict__ Ap,
                    const void* __restrict__ Bp,
                    float* __restrict__ Co,
                    bf16_t* __restrict__ Cb,
                    const float* __restrict__ bias,
                    const float* __restrict__ res,
                    int M, int N, int K, int lda, int ldb,
                    long long sAz, long long sBz, long long sCz,
                    int nH, long long sAh, long long sBh)
{
    // padded stride (40 bf16 = 80 B): 16B-aligned chunks, spread banks
    __shared__ bf16_t As[2][128][40];
    __shared__ bf16_t Bs[2][128][40];

    const int tid   = threadIdx.x;
    const int lane  = tid & 31;
    const int wave  = tid >> 5;
    const int waveM = wave & 3;    // 0..3 -> 32 rows each
    const int waveN = wave >> 2;   // 0..1 -> 64 cols each
    const int khalf = lane >> 4;   // lane half (ISA 7.12.2 layouts)
    const int lidx  = lane & 15;

    const int blockRow = blockIdx.y * 128;
    const int blockCol = blockIdx.x * 128;
    const long long z = blockIdx.z;

    // staging coordinates: row (M or N index), 16-wide K half
    const int srow = tid >> 1;
    const int scol = (tid & 1) * 16;

    const int kPerH = K >> 5;

    v8bf aR[2], bR[2];

    auto gload = [&](int h, int k0) {
        const long long aoff = z * sAz + (long long)h * sAh;
        const long long boff = z * sBz + (long long)h * sBh;
        const bf16_t* asrc = Ap + aoff + (long long)(blockRow + srow) * lda + k0 + scol;
        aR[0] = *(const v8bf*)asrc;
        aR[1] = *(const v8bf*)(asrc + 8);
        if (BMODE == B_BF16_NK) {
            const bf16_t* bsrc = (const bf16_t*)Bp + boff
                               + (long long)(blockCol + srow) * ldb + k0 + scol;
            bR[0] = *(const v8bf*)bsrc;
            bR[1] = *(const v8bf*)(bsrc + 8);
        } else { // B_F32_NK: fp32 weights, convert while staging
            const float* bsrc = (const float*)Bp + boff
                              + (long long)(blockCol + srow) * ldb + k0 + scol;
            #pragma unroll
            for (int q = 0; q < 4; ++q) {
                v4f f = *(const v4f*)(bsrc + 4 * q);
                #pragma unroll
                for (int j = 0; j < 4; ++j) {
                    int e = 4 * q + j;
                    if (e < 8) bR[0][e] = (bf16_t)f[j];
                    else       bR[1][e - 8] = (bf16_t)f[j];
                }
            }
        }
    };

    auto lstore = [&](int buf) {
        *(v8bf*)&As[buf][srow][scol]     = aR[0];
        *(v8bf*)&As[buf][srow][scol + 8] = aR[1];
        *(v8bf*)&Bs[buf][srow][scol]     = bR[0];
        *(v8bf*)&Bs[buf][srow][scol + 8] = bR[1];
    };

    v8f acc[2][4] = {};

    // ---- software pipeline: prefetch(next) || wmma(cur), 1 barrier/step ----
    gload(0, 0);
    lstore(0);
    __syncthreads();
    int cur = 0;
    for (int h = 0; h < nH; ++h) {
        for (int kk = 0; kk < kPerH; ++kk) {
            int kn = kk + 1, hn = h;
            bool has_next = true;
            if (kn == kPerH) { kn = 0; ++hn; if (hn == nH) has_next = false; }
            if (has_next) gload(hn, kn << 5);

            // fragments per ISA 7.12.2 bf16 layouts
            v16bf afrag[2], bfrag[4];
            #pragma unroll
            for (int mi = 0; mi < 2; ++mi) {
                int r = waveM * 32 + mi * 16 + lidx;              // M = lane&15
                v8bf lo = *(const v8bf*)&As[cur][r][khalf * 8];   // K = half*8 + i
                v8bf hi = *(const v8bf*)&As[cur][r][16 + khalf * 8];
                afrag[mi] = __builtin_shufflevector(lo, hi,
                    0,1,2,3,4,5,6,7,8,9,10,11,12,13,14,15);
            }
            #pragma unroll
            for (int ni = 0; ni < 4; ++ni) {
                int n = waveN * 64 + ni * 16 + lidx;              // N = lane&15
                v8bf lo = *(const v8bf*)&Bs[cur][n][khalf * 8];   // K = half*8 + i
                v8bf hi = *(const v8bf*)&Bs[cur][n][16 + khalf * 8];
                bfrag[ni] = __builtin_shufflevector(lo, hi,
                    0,1,2,3,4,5,6,7,8,9,10,11,12,13,14,15);
            }
            #pragma unroll
            for (int mi = 0; mi < 2; ++mi)
                #pragma unroll
                for (int ni = 0; ni < 4; ++ni)
                    acc[mi][ni] = __builtin_amdgcn_wmma_f32_16x16x32_bf16(
                        false, afrag[mi], false, bfrag[ni],
                        (short)0, acc[mi][ni], false, false);

            if (has_next) lstore(cur ^ 1);
            __syncthreads();
            cur ^= 1;
        }
    }

    // ---- epilogue (C layout: VGPR v, lanes0-15 M=v; lanes16-31 M=v+8) ----
    const long long zc = z * sCz;
    #pragma unroll
    for (int mi = 0; mi < 2; ++mi) {
        #pragma unroll
        for (int ni = 0; ni < 4; ++ni) {
            int rbase = blockRow + waveM * 32 + mi * 16 + khalf * 8;
            int c     = blockCol + waveN * 64 + ni * 16 + lidx;
            if (EPI == EPI_BF16T) {
                // 8 consecutive rows, fixed col -> contiguous 16B chunk in [N,M]
                v8bf p;
                #pragma unroll
                for (int v = 0; v < 8; ++v) p[v] = (bf16_t)acc[mi][ni][v];
                *(v8bf*)&Cb[zc + (long long)c * M + rbase] = p;
            } else {
                #pragma unroll
                for (int v = 0; v < 8; ++v) {
                    int r = rbase + v;
                    float val = acc[mi][ni][v];
                    long long idx = zc + (long long)r * N + c;
                    if (EPI == EPI_BF16) {
                        Cb[idx] = (bf16_t)val;
                    } else if (EPI == EPI_F32) {
                        Co[idx] = val;
                    } else if (EPI == EPI_RELU_BIAS_BF16) {
                        float x = val + bias[c];
                        Cb[idx] = (bf16_t)fmaxf(x, 0.0f);
                    } else if (EPI == EPI_RES_DUAL) {
                        float x = val + res[idx];
                        Co[idx] = x;
                        Cb[idx] = (bf16_t)x;
                    } else { // EPI_BIAS_RES_DUAL
                        float x = val + bias[c] + res[idx];
                        Co[idx] = x;
                        Cb[idx] = (bf16_t)x;
                    }
                }
            }
        }
    }
}

// Row softmax over width=2048 (one block per row), bf16 output.
__global__ __launch_bounds__(256)
void softmax_rows(const float* __restrict__ S, bf16_t* __restrict__ W,
                  int width, float lam)
{
    __shared__ float red[256];
    const int tid = threadIdx.x;
    const long long row = blockIdx.x;
    const float* src = S + row * (long long)width;

    float v[8];
    #pragma unroll
    for (int i = 0; i < 8; ++i) v[i] = lam * src[tid + i * 256];
    float m = v[0];
    #pragma unroll
    for (int i = 1; i < 8; ++i) m = fmaxf(m, v[i]);
    red[tid] = m; __syncthreads();
    for (int s = 128; s > 0; s >>= 1) {
        if (tid < s) red[tid] = fmaxf(red[tid], red[tid + s]);
        __syncthreads();
    }
    m = red[0]; __syncthreads();

    float e[8], sum = 0.f;
    #pragma unroll
    for (int i = 0; i < 8; ++i) { e[i] = __expf(v[i] - m); sum += e[i]; }
    red[tid] = sum; __syncthreads();
    for (int s = 128; s > 0; s >>= 1) {
        if (tid < s) red[tid] += red[tid + s];
        __syncthreads();
    }
    float inv = 1.0f / red[0];
    bf16_t* dst = W + row * (long long)width;
    #pragma unroll
    for (int i = 0; i < 8; ++i) dst[tid + i * 256] = (bf16_t)(e[i] * inv);
}

// Tiled transpose: src fp32 [R,C] -> dstF fp32 [C,R] (+ optional bf16 copy).
__global__ __launch_bounds__(256)
void transpose_cvt(const float* __restrict__ src, float* __restrict__ dstF,
                   bf16_t* __restrict__ dstB, int R, int C)
{
    __shared__ float tile[32][33];
    int x = blockIdx.x * 32 + threadIdx.x;
    int y = blockIdx.y * 32 + threadIdx.y;
    #pragma unroll
    for (int j = 0; j < 32; j += 8)
        tile[threadIdx.y + j][threadIdx.x] = src[(long long)(y + j) * C + x];
    __syncthreads();
    int x2 = blockIdx.y * 32 + threadIdx.x;
    int y2 = blockIdx.x * 32 + threadIdx.y;
    #pragma unroll
    for (int j = 0; j < 32; j += 8) {
        float v = tile[threadIdx.x][threadIdx.y + j];
        long long idx = (long long)(y2 + j) * R + x2;
        if (dstF) dstF[idx] = v;
        if (dstB) dstB[idx] = (bf16_t)v;
    }
}

extern "C" void kernel_launch(void* const* d_in, const int* in_sizes, int n_in,
                              void* d_out, int out_size, void* d_ws, size_t ws_size,
                              hipStream_t stream)
{
    (void)in_sizes; (void)n_in; (void)out_size; (void)ws_size;

    const float* X   = (const float*)d_in[0];
    const float* Qs  = (const float*)d_in[1];
    const float* Ks  = (const float*)d_in[2];
    const float* Vs  = (const float*)d_in[3];
    const float* W1s = (const float*)d_in[4];
    const float* b1s = (const float*)d_in[5];
    const float* W2s = (const float*)d_in[6];
    const float* b2s = (const float*)d_in[7];

    const int D = 1024, N = 2048, DFF = 4096, L = 10, H = 3;
    const long long DN = (long long)D * N;
    const long long NN = (long long)N * N;
    const long long ND = (long long)N * D;

    // ---- carve workspace (~150 MB), token-major "T" layouts ----
    char* w = (char*)d_ws;
    auto carve = [&](long long bytes) {
        char* p = w; w += (bytes + 255) & ~255LL; return p;
    };
    float*  XTf    = (float*) carve(ND * 4);           // X^T fp32 [n][d]
    bf16_t* Xt     = (bf16_t*)carve(ND * 2);           // X^T bf16
    bf16_t* QXt    = (bf16_t*)carve(H * ND * 2);       // [h][n][d]
    bf16_t* KXt    = (bf16_t*)carve(H * ND * 2);       // [h][n][d]
    bf16_t* VX     = (bf16_t*)carve(H * DN * 2);       // [h][d][n]  (feature-major)
    float*  St     = (float*) carve(H * NN * 4);       // scores^T [h][m][n]
    bf16_t* WT     = (bf16_t*)carve(H * NN * 2);       // attn_w^T bf16
    float*  attnTf = (float*) carve(ND * 4);           // attn^T fp32 [n][d]
    bf16_t* attnTb = (bf16_t*)carve(ND * 2);
    bf16_t* ff1T   = (bf16_t*)carve((long long)N * DFF * 2); // [n][f]

    dim3 blk(256);
    dim3 tblk(32, 8);

    // layer-0 input: X [d][n] -> XTf/Xt [n][d]
    transpose_cvt<<<dim3(N / 32, D / 32), tblk, 0, stream>>>(X, XTf, Xt, D, N);

    for (int l = 0; l < L; ++l) {
        const long long wqkv = (long long)l * H * D * D;

        // QXt_h[n][i] = sum_j Xt[n][j] * Qw_h[i][j]   (weights already B^T layout)
        dim3 gQK(D / 128, N / 128, H);
        gemm_wmma_bf16<B_F32_NK, EPI_BF16><<<gQK, blk, 0, stream>>>(
            Xt, Qs + wqkv, nullptr, QXt, nullptr, nullptr,
            N, D, D, D, D, 0, (long long)D * D, ND, 1, 0, 0);
        gemm_wmma_bf16<B_F32_NK, EPI_BF16><<<gQK, blk, 0, stream>>>(
            Xt, Ks + wqkv, nullptr, KXt, nullptr, nullptr,
            N, D, D, D, D, 0, (long long)D * D, ND, 1, 0, 0);
        // VX_h stored feature-major [d][n] via transposed chunk epilogue
        gemm_wmma_bf16<B_F32_NK, EPI_BF16T><<<gQK, blk, 0, stream>>>(
            Xt, Vs + wqkv, nullptr, VX, nullptr, nullptr,
            N, D, D, D, D, 0, (long long)D * D, DN, 1, 0, 0);

        // St[h][m][n] = sum_d QXt[m][d] * KXt[n][d]   (pure A*B^T, all direct)
        dim3 gS(N / 128, N / 128, H);
        gemm_wmma_bf16<B_BF16_NK, EPI_F32><<<gS, blk, 0, stream>>>(
            QXt, KXt, St, nullptr, nullptr, nullptr,
            N, N, D, D, D, ND, ND, NN, 1, 0, 0);

        // softmax over reference axis n == row axis of St
        softmax_rows<<<dim3(H * N), blk, 0, stream>>>(St, WT, N, 1.0f);

        // attnT[m][d] = XTf + sum_h sum_n WT_h[m][n] * VX_h[d][n]
        dim3 gA(D / 128, N / 128, 1);
        gemm_wmma_bf16<B_BF16_NK, EPI_RES_DUAL><<<gA, blk, 0, stream>>>(
            WT, VX, attnTf, attnTb, nullptr, XTf,
            N, D, N, N, N, 0, 0, 0, H, NN, DN);

        // ff1T[n][f] = relu(sum_d attnT[n][d] * W1[f][d] + b1[f])
        dim3 gF1(DFF / 128, N / 128, 1);
        gemm_wmma_bf16<B_F32_NK, EPI_RELU_BIAS_BF16><<<gF1, blk, 0, stream>>>(
            attnTb, W1s + (long long)l * DFF * D, nullptr, ff1T,
            b1s + (long long)l * DFF, nullptr,
            N, DFF, D, D, D, 0, 0, 0, 1, 0, 0);

        // XT_next[n][i] = attnT + sum_f ff1T[n][f] * W2[i][f] + b2[i]
        dim3 gF2(D / 128, N / 128, 1);
        gemm_wmma_bf16<B_F32_NK, EPI_BIAS_RES_DUAL><<<gF2, blk, 0, stream>>>(
            ff1T, W2s + (long long)l * D * DFF, XTf, Xt,
            b2s + (long long)l * D, attnTf,
            N, D, DFF, DFF, DFF, 0, 0, 0, 1, 0, 0);
    }

    // final: XTf [n][d] -> d_out [d][n]
    transpose_cvt<<<dim3(D / 32, N / 32), tblk, 0, stream>>>(
        XTf, (float*)d_out, nullptr, N, D);
}
